// PromptQueryAdapter_19585050869953
// MI455X (gfx1250) — matrix operand order repaired and checked by reading, not compiled, not run on hardware
//
#include <hip/hip_runtime.h>
#include <cstdint>
#include <cstddef>

typedef _Float16 h16;
typedef __attribute__((ext_vector_type(16))) _Float16 v16h;
typedef __attribute__((ext_vector_type(8)))  _Float16 v8h;
typedef __attribute__((ext_vector_type(8)))  float    v8f;

constexpr int cL = 2, cB = 32, cN = 225, cM = 1800, cD = 768, cH = 256, cMID = 128;
constexpr int cGS = 15, cIMG = 240, cTOPK = 10;
constexpr int cNPAD = 256;   // rows padded to block-row (32) multiple: 8 row-tiles
constexpr int cMPAD = 1856;  // cols padded to 64 multiple: 29 col-groups
constexpr float cBETA = 1.0f, cGR = 0.5f, cGC = 0.5f, cEPS = 1e-5f;

// ---------------------------------------------------------------------------
// 1) L2-normalize rows of [B, rowsValid, D] f32 -> [B, rowsPad, D] f16 (pads = 0)
// ---------------------------------------------------------------------------
__global__ void k_l2norm_f16(const float* __restrict__ x, h16* __restrict__ y,
                             int rowsValid, int rowsPad, int d) {
  int row = blockIdx.x;                 // over B*rowsPad
  int b = row / rowsPad;
  int r = row % rowsPad;
  h16* yr = y + (size_t)row * d;
  if (r >= rowsValid) {
    for (int i = threadIdx.x; i < d; i += blockDim.x) yr[i] = (h16)0.f;
    return;
  }
  const float* xr = x + ((size_t)b * rowsValid + r) * d;
  __shared__ float red[256];
  float s = 0.f;
  for (int i = threadIdx.x; i < d; i += blockDim.x) { float v = xr[i]; s += v * v; }
  red[threadIdx.x] = s; __syncthreads();
  for (int off = 128; off > 0; off >>= 1) {
    if ((int)threadIdx.x < off) red[threadIdx.x] += red[threadIdx.x + off];
    __syncthreads();
  }
  float inv = 1.f / fmaxf(sqrtf(red[0]), 1e-12f);
  for (int i = threadIdx.x; i < d; i += blockDim.x) yr[i] = (h16)(xr[i] * inv);
}

// ---------------------------------------------------------------------------
// 2) Batched NT WMMA GEMM, register-blocked + software-pipelined:
//    C[r,c] = sum_k A[r,k]*Bm[c,k]
//    A : [rowsPad, K] f16, rowsPad % 32 == 0 (zero padded by producer)
//    Bm: [colsPad, K] f16, colsPad % 64 == 0 (zero padded by producer)
//    C : [rowsPad, colsPad] f32;  K % 64 == 0.
//    Block = 256 threads = 8 waves = 2 row-tiles x 4 col-groups.
//    Block tile = 32 rows x 256 cols; wave tile = 16 rows x 64 cols
//    (4 accumulators, A fragment reused 4x).
//    Per 64-k round: ds_store the pre-fetched A panel into one of two LDS
//    buffers, single barrier, issue next round's global A fetch (stays in
//    flight under this round's 8 WMMAs + 16 B-loads).
// ---------------------------------------------------------------------------
__global__ void k_wmma_gemm_nt(const h16* __restrict__ A, const h16* __restrict__ Bm,
                               float* __restrict__ C, int K, int colsPad,
                               long long sAb, long long sBb, long long sCb) {
  const int batch = blockIdx.z;
  const h16* Ab = A + (long long)batch * sAb;
  const h16* Bb = Bm + (long long)batch * sBb;
  float* Cb = C + (long long)batch * sCb;

  const int wave = threadIdx.x >> 5;
  const int lane = threadIdx.x & 31;
  const int waveRow = wave >> 2;           // 0..1
  const int waveCol = wave & 3;            // 0..3
  const int r0b = blockIdx.y * 32;         // block row base
  const int rw = waveRow * 16;             // wave row offset within block
  const int c0 = (blockIdx.x * 4 + waveCol) * 64;  // wave col base
  const bool active = (c0 < colsPad);      // wave-uniform (colsPad % 64 == 0)
  const int rowA = lane & 15;
  const int sel = lane >> 4;

  __shared__ __align__(32) h16 As[2][32 * 64];   // double-buffered 4 KB A panels

  // this thread's staging slot: one b128 of the 32x64 panel per round
  const int tid = (int)threadIdx.x;
  const int rr = tid >> 3;                 // 0..31 (panel row)
  const int kst = (tid & 7) * 8;           // 0..56 (panel k offset)
  const h16* Ag = Ab + (size_t)(r0b + rr) * K + kst;

  // per-wave B base pointers: column = c0 + j*16 + (lane&15), contiguous in k
  const h16* Bp0 = Bb + (size_t)(c0 + 0 * 16 + (lane & 15)) * K + (sel << 4);
  const h16* Bp1 = Bb + (size_t)(c0 + 1 * 16 + (lane & 15)) * K + (sel << 4);
  const h16* Bp2 = Bb + (size_t)(c0 + 2 * 16 + (lane & 15)) * K + (sel << 4);
  const h16* Bp3 = Bb + (size_t)(c0 + 3 * 16 + (lane & 15)) * K + (sel << 4);

  v8f acc0 = {}, acc1 = {}, acc2 = {}, acc3 = {};

  v8h areg = *(const v8h*)Ag;              // prologue: round-0 A chunk

  for (int k0 = 0; k0 < K; k0 += 64) {
    const int buf = (k0 >> 6) & 1;
    *(v8h*)&As[buf][rr * 64 + kst] = areg;
    __syncthreads();
    if (k0 + 64 < K) {
      areg = *(const v8h*)(Ag + k0 + 64);  // next round's A, in flight now
    }
    if (active) {
      if (k0 + 64 < K) {                   // prefetch next-round B stream
        __builtin_prefetch((const void*)(Bp0 + k0 + 64), 0, 1);
        __builtin_prefetch((const void*)(Bp2 + k0 + 64), 0, 1);
      }
#pragma unroll
      for (int ks = 0; ks < 2; ++ks) {
        // A fragment, contiguous k order (matches B):
        //  lanes 0-15: row=lane,    K in {0..15} of this 32-k step
        //  lanes 16-31: row=lane-16, K in {16..31}
        const int abase = (rw + rowA) * 64 + ks * 32 + sel * 16;
        v8h lo = *(const v8h*)&As[buf][abase];
        v8h hi = *(const v8h*)&As[buf][abase + 8];
        v16h a;
#pragma unroll
        for (int i = 0; i < 8; ++i) { a[i] = lo[i]; a[i + 8] = hi[i]; }
        const int kk = k0 + ks * 32;
        v16h b0 = *(const v16h*)(Bp0 + kk);
        v16h b1 = *(const v16h*)(Bp1 + kk);
        v16h b2 = *(const v16h*)(Bp2 + kk);
        v16h b3 = *(const v16h*)(Bp3 + kk);
        acc0 = __builtin_amdgcn_wmma_f32_16x16x32_f16(false, a, false, b0, (short)0, acc0, false, false);
        acc1 = __builtin_amdgcn_wmma_f32_16x16x32_f16(false, a, false, b1, (short)0, acc1, false, false);
        acc2 = __builtin_amdgcn_wmma_f32_16x16x32_f16(false, a, false, b2, (short)0, acc2, false, false);
        acc3 = __builtin_amdgcn_wmma_f32_16x16x32_f16(false, a, false, b3, (short)0, acc3, false, false);
      }
    }
    // no second barrier needed: next round's ds_store hits the other panel,
    // and this round's ds_loads are consumed (DScnt-waited) before any wave
    // can signal the next round's barrier.
  }

  if (active) {
    const int col = lane & 15;
    const int rbase = r0b + rw + sel * 8;
#pragma unroll
    for (int v = 0; v < 8; ++v) {
      float* cr = Cb + (size_t)(rbase + v) * colsPad + c0 + col;
      cr[0] = acc0[v];
      cr[16] = acc1[v];
      cr[32] = acc2[v];
      cr[48] = acc3[v];
    }
  }
}

// ---------------------------------------------------------------------------
// 3) row max / argmax over the similarity matrix
// ---------------------------------------------------------------------------
__global__ void k_argmax(const float* __restrict__ sim, float* __restrict__ outResid,
                         float* __restrict__ outIdxF, int* __restrict__ idxI,
                         int mvalid, int mpad) {
  int b = blockIdx.x / cN, n = blockIdx.x % cN;
  const float* row = sim + ((size_t)b * cNPAD + n) * mpad;
  __shared__ float sv[256];
  __shared__ int si[256];
  float best = -1e30f; int bi = 0;
  for (int m = threadIdx.x; m < mvalid; m += blockDim.x) {
    float v = row[m];
    if (v > best) { best = v; bi = m; }
  }
  sv[threadIdx.x] = best; si[threadIdx.x] = bi; __syncthreads();
  for (int off = 128; off > 0; off >>= 1) {
    if ((int)threadIdx.x < off) {
      float vo = sv[threadIdx.x + off];
      if (vo > sv[threadIdx.x] ||
          (vo == sv[threadIdx.x] && si[threadIdx.x + off] < si[threadIdx.x])) {
        sv[threadIdx.x] = vo; si[threadIdx.x] = si[threadIdx.x + off];
      }
    }
    __syncthreads();
  }
  if (threadIdx.x == 0) {
    idxI[blockIdx.x] = si[0];
    outResid[blockIdx.x] = 1.f - sv[0];
    outIdxF[blockIdx.x] = (float)si[0];
  }
}

// ---------------------------------------------------------------------------
// 4) ctx = q + BETA*|q - p[idx]|, transpose to [B,D,225], eval-BN; f32 + f16 copies
// ---------------------------------------------------------------------------
__global__ void k_cmap(const float* __restrict__ q, const float* __restrict__ p,
                       const int* __restrict__ idx, const float* __restrict__ g,
                       const float* __restrict__ bta, const float* __restrict__ mm,
                       const float* __restrict__ vv, float* __restrict__ cmap32,
                       h16* __restrict__ cmap16) {
  int bn = blockIdx.x;                    // B*N
  int b = bn / cN, n = bn % cN;
  const float* qr = q + ((size_t)b * cN + n) * cD;
  const float* pr = p + ((size_t)b * cM + idx[bn]) * cD;
  for (int d = threadIdx.x; d < cD; d += blockDim.x) {
    float qv = qr[d];
    float ctx = qv + cBETA * fabsf(qv - pr[d]);
    float y = (ctx - mm[d]) * (g[d] * rsqrtf(vv[d] + cEPS)) + bta[d];
    size_t o = ((size_t)b * cD + d) * cN + n;
    cmap32[o] = y; cmap16[o] = (h16)y;
  }
}

// ---------------------------------------------------------------------------
// 5) f32 -> f16 convert (weights)
// ---------------------------------------------------------------------------
__global__ void k_f32_to_f16(const float* __restrict__ x, h16* __restrict__ y, int n) {
  int i = blockIdx.x * blockDim.x + threadIdx.x;
  if (i < n) y[i] = (h16)x[i];
}

// ---------------------------------------------------------------------------
// 6) im2col for 3x3 pad-1 conv: x [B,C,S,S] f16 -> col [B*S*S, C*9] f16
// ---------------------------------------------------------------------------
__global__ void k_im2col3(const h16* __restrict__ x, h16* __restrict__ col,
                          int C, int S) {
  long long i = (long long)blockIdx.x * blockDim.x + threadIdx.x;
  long long total = (long long)cB * S * S * C * 9;
  if (i >= total) return;
  int Kc = C * 9;
  long long row = i / Kc;
  int k = (int)(i % Kc);
  int c = k / 9, kk = k % 9;
  int ky = kk / 3 - 1, kx = kk % 3 - 1;
  int b = (int)(row / (S * S));
  int pxy = (int)(row % (S * S));
  int y0 = pxy / S, x0 = pxy % S;
  int yy = y0 + ky, xx = x0 + kx;
  h16 v = (h16)0.f;
  if (yy >= 0 && yy < S && xx >= 0 && xx < S)
    v = x[(((size_t)b * C + c) * S + yy) * S + xx];
  col[i] = v;
}

// ---------------------------------------------------------------------------
// 7) BN + ReLU, GEMM-row layout [B*S*S, O] f32 -> NCHW [B,O,S,S] f16
// ---------------------------------------------------------------------------
__global__ void k_bnrelu_nchw(const float* __restrict__ Gm, h16* __restrict__ X,
                              const float* __restrict__ g, const float* __restrict__ bta,
                              const float* __restrict__ mm, const float* __restrict__ vv,
                              int O, int S) {
  long long i = (long long)blockIdx.x * blockDim.x + threadIdx.x;
  long long total = (long long)cB * O * S * S;
  if (i >= total) return;
  int xw = (int)(i % S); long long t = i / S;
  int yw = (int)(t % S); t /= S;
  int o = (int)(t % O); int b = (int)(t / O);
  size_t row = ((size_t)b * S + yw) * S + xw;
  float v = Gm[row * O + o];
  v = (v - mm[o]) * (g[o] * rsqrtf(vv[o] + cEPS)) + bta[o];
  X[i] = (h16)fmaxf(v, 0.f);
}

// ---------------------------------------------------------------------------
// 8) ConvTranspose2d kernel=2 stride=2: X [B,Cin,S,S] f16, W [Cin,Cout,2,2] f32
// ---------------------------------------------------------------------------
__global__ void k_convt2(const h16* __restrict__ X, const float* __restrict__ W,
                         const float* __restrict__ bias, h16* __restrict__ Y,
                         int Cin, int Cout, int S) {
  long long i = (long long)blockIdx.x * blockDim.x + threadIdx.x;
  int S2 = 2 * S;
  long long total = (long long)cB * Cout * S2 * S2;
  if (i >= total) return;
  int ox = (int)(i % S2); long long t = i / S2;
  int oy = (int)(t % S2); t /= S2;
  int o = (int)(t % Cout); int b = (int)(t / Cout);
  int hx = ox >> 1, kx = ox & 1, hy = oy >> 1, ky = oy & 1;
  const h16* xp = X + ((size_t)b * Cin * S + hy) * S + hx;
  const float* wp = W + (size_t)o * 4 + ky * 2 + kx;
  float s = 0.f;
  for (int c = 0; c < Cin; ++c)
    s += (float)xp[(size_t)c * S * S] * wp[(size_t)c * Cout * 4];
  Y[i] = (h16)(s + bias[o]);
}

// ---------------------------------------------------------------------------
// 9) 1x1 conv to 2 channels + bias -> f32 (local2c, straight into d_out)
// ---------------------------------------------------------------------------
__global__ void k_conv1x1(const h16* __restrict__ X, const float* __restrict__ W,
                          const float* __restrict__ bias, float* __restrict__ Y,
                          int Cin, int P) {
  long long i = (long long)blockIdx.x * blockDim.x + threadIdx.x;
  long long total = (long long)cB * 2 * P;
  if (i >= total) return;
  int p = (int)(i % P); long long t = i / P;
  int ch = (int)(t % 2); int b = (int)(t / 2);
  const h16* xp = X + (size_t)b * Cin * P + p;
  const float* wp = W + (size_t)ch * Cin;
  float s = 0.f;
  for (int c = 0; c < Cin; ++c) s += (float)xp[(size_t)c * P] * wp[c];
  Y[i] = s + bias[ch];
}

// ---------------------------------------------------------------------------
// Anti-aliased separable triangle resize (matches jax.image.resize 'bilinear'
// with per-output weight normalization); degrades to plain bilinear on upscale.
// ---------------------------------------------------------------------------
__device__ inline void resize_tri2(const float* __restrict__ src, int Sin, int Sout,
                                   int y, int x, size_t chStride, float* out2) {
  float scale = (float)Sin / (float)Sout;
  float ks = scale > 1.f ? scale : 1.f;
  float cy = (y + 0.5f) * scale - 0.5f;
  float cx = (x + 0.5f) * scale - 0.5f;
  int y0 = (int)ceilf(cy - ks), y1 = (int)floorf(cy + ks);
  int x0 = (int)ceilf(cx - ks), x1 = (int)floorf(cx + ks);
  float a0 = 0.f, a1 = 0.f, den = 0.f;
  for (int yi = y0; yi <= y1; ++yi) {
    if (yi < 0 || yi >= Sin) continue;
    float wy = 1.f - fabsf((float)yi - cy) / ks;
    if (wy <= 0.f) continue;
    for (int xi = x0; xi <= x1; ++xi) {
      if (xi < 0 || xi >= Sin) continue;
      float wx = 1.f - fabsf((float)xi - cx) / ks;
      if (wx <= 0.f) continue;
      float w = wy * wx;
      a0 += w * src[yi * Sin + xi];
      a1 += w * src[chStride + yi * Sin + xi];
      den += w;
    }
  }
  float inv = 1.f / fmaxf(den, 1e-20f);
  out2[0] = a0 * inv; out2[1] = a1 * inv;
}

// 10) resize 60->240 + channel softmax -> local_score
__global__ void k_resize_softmax(const float* __restrict__ Lc, float* __restrict__ Sc) {
  long long i = (long long)blockIdx.x * blockDim.x + threadIdx.x;
  long long total = (long long)cB * cIMG * cIMG;
  if (i >= total) return;
  int x = (int)(i % cIMG); long long t = i / cIMG;
  int y = (int)(t % cIMG); int b = (int)(t / cIMG);
  const float* src = Lc + (size_t)b * 2 * 3600;
  float c2[2];
  resize_tri2(src, 60, cIMG, y, x, 3600, c2);
  float mx = fmaxf(c2[0], c2[1]);
  float e0 = __expf(c2[0] - mx), e1 = __expf(c2[1] - mx);
  float inv = 1.f / (e0 + e1);
  size_t base = (size_t)b * 2 * cIMG * cIMG + (size_t)y * cIMG + x;
  Sc[base] = e0 * inv;
  Sc[base + (size_t)cIMG * cIMG] = e1 * inv;
}

// 11) resize 60->15 + patch head: patch_logit, evidence
__global__ void k_patch(const float* __restrict__ Lc, const float* __restrict__ resid,
                        float* __restrict__ evid, float* __restrict__ plog) {
  int i = blockIdx.x * blockDim.x + threadIdx.x;
  if (i >= cB * cN) return;
  int b = i / cN, n = i % cN;
  int y = n / cGS, x = n % cGS;
  const float* src = Lc + (size_t)b * 2 * 3600;
  float c2[2];
  resize_tri2(src, 60, cGS, y, x, 3600, c2);
  plog[i] = c2[1] - c2[0];
  float mx = fmaxf(c2[0], c2[1]);
  float e0 = __expf(c2[0] - mx), e1 = __expf(c2[1] - mx);
  float ps1 = e1 / (e0 + e1);
  evid[i] = cGR * resid[i] + cGC * ps1;
}

// 12) gfeat: 0.5*(mean + top-10 mean) over [B,D,225]
__global__ void k_gfeat(const float* __restrict__ cmap, float* __restrict__ gf) {
  int row = blockIdx.x;                   // b*D + d
  const float* src = cmap + (size_t)row * cN;
  __shared__ float vals[256];
  __shared__ float rv[256];
  __shared__ int ri[256];
  int t = threadIdx.x;
  vals[t] = (t < cN) ? src[t] : -1e30f;
  __syncthreads();
  rv[t] = (t < cN) ? vals[t] : 0.f;
  __syncthreads();
  for (int off = 128; off > 0; off >>= 1) {
    if (t < off) rv[t] += rv[t + off];
    __syncthreads();
  }
  float mean = rv[0] / (float)cN;
  __syncthreads();
  float topsum = 0.f;
  for (int it = 0; it < cTOPK; ++it) {
    rv[t] = vals[t]; ri[t] = t; __syncthreads();
    for (int off = 128; off > 0; off >>= 1) {
      if (t < off && rv[t + off] > rv[t]) { rv[t] = rv[t + off]; ri[t] = ri[t + off]; }
      __syncthreads();
    }
    if (t == 0) { topsum += rv[0]; vals[ri[0]] = -1e30f; }
    __syncthreads();
  }
  if (t == 0) gf[row] = 0.5f * (mean + topsum / (float)cTOPK);
}

// 13) FC (+optional BN+ReLU): Y[b,o] = f(X[b,:] . W[o,:])
__global__ void k_fc(const float* __restrict__ X, const float* __restrict__ W,
                     const float* __restrict__ g, const float* __restrict__ bta,
                     const float* __restrict__ mm, const float* __restrict__ vv,
                     float* __restrict__ Y, int Kd, int O, int bnrelu) {
  int b = blockIdx.x / O, o = blockIdx.x % O;
  const float* xr = X + (size_t)b * Kd;
  const float* wr = W + (size_t)o * Kd;
  __shared__ float red[256];
  float s = 0.f;
  for (int k = threadIdx.x; k < Kd; k += blockDim.x) s += xr[k] * wr[k];
  red[threadIdx.x] = s; __syncthreads();
  for (int off = 128; off > 0; off >>= 1) {
    if ((int)threadIdx.x < off) red[threadIdx.x] += red[threadIdx.x + off];
    __syncthreads();
  }
  if (threadIdx.x == 0) {
    float v = red[0];
    if (bnrelu)
      v = fmaxf((v - mm[o]) * (g[o] * rsqrtf(vv[o] + cEPS)) + bta[o], 0.f);
    Y[(size_t)b * O + o] = v;
  }
}

__global__ void k_glogit(const float* __restrict__ g2c, float* __restrict__ glog) {
  int b = threadIdx.x;
  if (b < cB) glog[b] = g2c[b * 2 + 1] - g2c[b * 2 + 0];
}

// ---------------------------------------------------------------------------
// Host launcher
// ---------------------------------------------------------------------------
static inline unsigned cdiv(long long a, long long b) { return (unsigned)((a + b - 1) / b); }

extern "C" void kernel_launch(void* const* d_in, const int* in_sizes, int n_in,
                              void* d_out, int out_size, void* d_ws, size_t ws_size,
                              hipStream_t stream) {
  (void)in_sizes; (void)n_in; (void)out_size; (void)ws_size;

  // ---- inputs -------------------------------------------------------------
  const float* in_query = (const float*)d_in[0];
  const float* in_prompt = (const float*)d_in[1];
  const float* in_bns_g = (const float*)d_in[2];
  const float* in_bns_b = (const float*)d_in[3];
  const float* in_bns_m = (const float*)d_in[4];
  const float* in_bns_v = (const float*)d_in[5];
  const float* in_lc1_w = (const float*)d_in[6];
  const float* in_lbn1_g = (const float*)d_in[7];
  const float* in_lbn1_b = (const float*)d_in[8];
  const float* in_lbn1_m = (const float*)d_in[9];
  const float* in_lbn1_v = (const float*)d_in[10];
  const float* in_lt1_w = (const float*)d_in[11];
  const float* in_lt1_b = (const float*)d_in[12];
  const float* in_lc2_w = (const float*)d_in[13];
  const float* in_lbn2_g = (const float*)d_in[14];
  const float* in_lbn2_b = (const float*)d_in[15];
  const float* in_lbn2_m = (const float*)d_in[16];
  const float* in_lbn2_v = (const float*)d_in[17];
  const float* in_lt2_w = (const float*)d_in[18];
  const float* in_lt2_b = (const float*)d_in[19];
  const float* in_lc3_w = (const float*)d_in[20];
  const float* in_lc3_b = (const float*)d_in[21];
  const float* in_g1_w = (const float*)d_in[22];
  const float* in_gbn1_g = (const float*)d_in[23];
  const float* in_gbn1_b = (const float*)d_in[24];
  const float* in_gbn1_m = (const float*)d_in[25];
  const float* in_gbn1_v = (const float*)d_in[26];
  const float* in_g2_w = (const float*)d_in[27];
  const float* in_gbn2_g = (const float*)d_in[28];
  const float* in_gbn2_b = (const float*)d_in[29];
  const float* in_gbn2_m = (const float*)d_in[30];
  const float* in_gbn2_v = (const float*)d_in[31];
  const float* in_g3_w = (const float*)d_in[32];

  // ---- workspace layout (bytes); COL aliases QN/PN (temporally disjoint) --
  constexpr size_t MB = 1ull << 20;
  uint8_t* ws = (uint8_t*)d_ws;
  h16* QN = (h16*)(ws + 0);                // B*256*768 f16        (12.6 MB)
  h16* PN = (h16*)(ws + 16 * MB);          // B*1856*768 f16       (91.2 MB)
  float* SIM = (float*)(ws + 112 * MB);    // B*256*1856 f32       (60.8 MB)
  h16* COL = (h16*)(ws + 0);               // alias, max 132.7 MB
  float* CMAP32 = (float*)(ws + 176 * MB); // 22.1 MB
  h16* CMAP16 = (h16*)(ws + 200 * MB);     // 11.1 MB
  h16* W16 = (h16*)(ws + 212 * MB);        // 3.5 MB
  float* GOUT = (float*)(ws + 216 * MB);   // 14.7 MB
  h16* X1 = (h16*)(ws + 232 * MB);         // 3.7 MB
  h16* UP1 = (h16*)(ws + 236 * MB);        // 14.7 MB
  h16* X2 = (h16*)(ws + 252 * MB);         // 7.4 MB
  h16* UP2 = (h16*)(ws + 260 * MB);        // 29.5 MB
  int* IDX = (int*)(ws + 292 * MB);
  float* GFEAT = (float*)(ws + 293 * MB);
  float* H1 = (float*)(ws + 294 * MB);
  float* H2 = (float*)(ws + 295 * MB);

  // ---- output layout (floats, concatenated in return order) ---------------
  float* out = (float*)d_out;
  float* o_evid = out + 0;             // [L,B,225]
  float* o_plog = out + 14400;         // [L,B,225]
  float* o_glog = out + 28800;         // [L,B]
  float* o_g2c = out + 28864;          // [L,B,2]
  float* o_l2c = out + 28992;          // [L,B,2,60,60]
  float* o_lsc = out + 489792;         // [L,B,2,240,240]
  float* o_res = out + 7862592;        // [L,B,225]
  float* o_idx = out + 7876992;        // [L,B,225]

  for (int l = 0; l < cL; ++l) {
    const float* q_l = in_query + (size_t)l * cB * cN * cD;
    const float* p_l = in_prompt + (size_t)l * cB * cM * cD;
    float* evid = o_evid + (size_t)l * cB * cN;
    float* plog = o_plog + (size_t)l * cB * cN;
    float* glog = o_glog + (size_t)l * cB;
    float* g2c = o_g2c + (size_t)l * cB * 2;
    float* l2c = o_l2c + (size_t)l * cB * 2 * 3600;
    float* lsc = o_lsc + (size_t)l * cB * 2 * cIMG * cIMG;
    float* res = o_res + (size_t)l * cB * cN;
    float* idxf = o_idx + (size_t)l * cB * cN;

    // 1) normalize q, p -> f16 (padded)
    k_l2norm_f16<<<cB * cNPAD, 256, 0, stream>>>(q_l, QN, cN, cNPAD, cD);
    k_l2norm_f16<<<cB * cMPAD, 256, 0, stream>>>(p_l, PN, cM, cMPAD, cD);

    // 2) cosine similarity GEMM (batched): SIM = QN * PN^T
    {
      dim3 g(cdiv(cMPAD / 64, 4), cNPAD / 32, cB);     // (8, 8, 32)
      k_wmma_gemm_nt<<<g, 256, 0, stream>>>(QN, PN, SIM, cD, cMPAD,
                                            (long long)cNPAD * cD,
                                            (long long)cMPAD * cD,
                                            (long long)cNPAD * cMPAD);
    }

    // 3) per-row max / argmax, residual, idx
    k_argmax<<<cB * cN, 256, 0, stream>>>(SIM, res, idxf, IDX, cM, cMPAD);

    // 4) ctx + BN -> cmap [B,D,15,15]
    k_cmap<<<cB * cN, 256, 0, stream>>>(q_l, p_l, IDX,
                                        in_bns_g + (size_t)l * cD, in_bns_b + (size_t)l * cD,
                                        in_bns_m + (size_t)l * cD, in_bns_v + (size_t)l * cD,
                                        CMAP32, CMAP16);

    // 5) conv1 (768->256, 3x3) via im2col + WMMA GEMM
    k_f32_to_f16<<<cdiv(cH * cD * 9, 256), 256, 0, stream>>>(
        in_lc1_w + (size_t)l * cH * cD * 9, W16, cH * cD * 9);
    k_im2col3<<<cdiv((long long)cB * cGS * cGS * cD * 9, 256), 256, 0, stream>>>(
        CMAP16, COL, cD, cGS);
    {
      dim3 g(1, (cB * cGS * cGS) / 32, 1);             // cols 256 = 4 groups
      k_wmma_gemm_nt<<<g, 256, 0, stream>>>(COL, W16, GOUT, cD * 9, cH, 0, 0, 0);
    }
    k_bnrelu_nchw<<<cdiv((long long)cB * cH * cGS * cGS, 256), 256, 0, stream>>>(
        GOUT, X1, in_lbn1_g + (size_t)l * cH, in_lbn1_b + (size_t)l * cH,
        in_lbn1_m + (size_t)l * cH, in_lbn1_v + (size_t)l * cH, cH, cGS);

    // 6) ConvT 2x2 -> [B,256,30,30]
    k_convt2<<<cdiv((long long)cB * cH * 900, 256), 256, 0, stream>>>(
        X1, in_lt1_w + (size_t)l * cH * cH * 4, in_lt1_b + (size_t)l * cH, UP1, cH, cH, cGS);

    // 7) conv2 (256->128, 3x3) via im2col + WMMA GEMM
    k_f32_to_f16<<<cdiv(cMID * cH * 9, 256), 256, 0, stream>>>(
        in_lc2_w + (size_t)l * cMID * cH * 9, W16, cMID * cH * 9);
    k_im2col3<<<cdiv((long long)cB * 30 * 30 * cH * 9, 256), 256, 0, stream>>>(
        UP1, COL, cH, 30);
    {
      dim3 g(1, (cB * 900) / 32, 1);                   // cols 128 = 2 groups
      k_wmma_gemm_nt<<<g, 256, 0, stream>>>(COL, W16, GOUT, cH * 9, cMID, 0, 0, 0);
    }
    k_bnrelu_nchw<<<cdiv((long long)cB * cMID * 900, 256), 256, 0, stream>>>(
        GOUT, X2, in_lbn2_g + (size_t)l * cMID, in_lbn2_b + (size_t)l * cMID,
        in_lbn2_m + (size_t)l * cMID, in_lbn2_v + (size_t)l * cMID, cMID, 30);

    // 8) ConvT 2x2 -> [B,128,60,60]
    k_convt2<<<cdiv((long long)cB * cMID * 3600, 256), 256, 0, stream>>>(
        X2, in_lt2_w + (size_t)l * cMID * cMID * 4, in_lt2_b + (size_t)l * cMID,
        UP2, cMID, cMID, 30);

    // 9) 1x1 conv -> local2c (d_out)
    k_conv1x1<<<cdiv((long long)cB * 2 * 3600, 256), 256, 0, stream>>>(
        UP2, in_lc3_w + (size_t)l * 2 * cMID, in_lc3_b + (size_t)l * 2, l2c, cMID, 3600);

    // 10) resize 60->240 + softmax -> local_score
    k_resize_softmax<<<cdiv((long long)cB * cIMG * cIMG, 256), 256, 0, stream>>>(l2c, lsc);

    // 11) resize 60->15 + patch head -> patch_logit, evidence
    k_patch<<<cdiv(cB * cN, 256), 256, 0, stream>>>(l2c, res, evid, plog);

    // 12) global head
    k_gfeat<<<cB * cD, 256, 0, stream>>>(CMAP32, GFEAT);
    k_fc<<<cB * cH, 256, 0, stream>>>(GFEAT, in_g1_w + (size_t)l * cH * cD,
                                      in_gbn1_g + (size_t)l * cH, in_gbn1_b + (size_t)l * cH,
                                      in_gbn1_m + (size_t)l * cH, in_gbn1_v + (size_t)l * cH,
                                      H1, cD, cH, 1);
    k_fc<<<cB * cMID, 256, 0, stream>>>(H1, in_g2_w + (size_t)l * cMID * cH,
                                        in_gbn2_g + (size_t)l * cMID, in_gbn2_b + (size_t)l * cMID,
                                        in_gbn2_m + (size_t)l * cMID, in_gbn2_v + (size_t)l * cMID,
                                        H2, cH, cMID, 1);
    k_fc<<<cB * 2, 256, 0, stream>>>(H2, in_g3_w + (size_t)l * 2 * cMID,
                                     nullptr, nullptr, nullptr, nullptr, g2c, cMID, 2, 0);
    k_glogit<<<1, 32, 0, stream>>>(g2c, glog);
  }
}
